// get_model_44349832298551
// MI455X (gfx1250) — matrix operand order repaired and checked by reading, not compiled, not run on hardware
//
#include <hip/hip_runtime.h>
#include <hip/hip_bf16.h>
#include <math.h>

#define EPS_BN 1e-5f

typedef __attribute__((ext_vector_type(16))) _Float16 v16h;
typedef __attribute__((ext_vector_type(8)))  _Float16 v8h;
typedef __attribute__((ext_vector_type(8)))  float    v8f;

// ------------------------------------------------------------------
// prep: xyv (B,4,N) -> l0_points (B,N,4), l0_xy (B,N,2)
// ------------------------------------------------------------------
__global__ void prep_kernel(const float* __restrict__ xyv, float* __restrict__ l0p,
                            float* __restrict__ l0xy, int B, int N)
{
    int t = blockIdx.x * blockDim.x + threadIdx.x;
    int total = B * N;
    if (t >= total) return;
    int b = t / N, n = t % N;
    const float* src = xyv + (size_t)b * 4 * N + n;
    float v0 = src[0], v1 = src[(size_t)N], v2 = src[(size_t)2 * N], v3 = src[(size_t)3 * N];
    float* d = l0p + (size_t)t * 4;
    d[0] = v0; d[1] = v1; d[2] = v2; d[3] = v3;
    l0xy[(size_t)t * 2]     = v0;
    l0xy[(size_t)t * 2 + 1] = v1;
}

// ------------------------------------------------------------------
// Farthest point sampling: one block per batch, sequential like the
// reference lax.scan (emits `far` BEFORE the update; far0 = 0).
// ------------------------------------------------------------------
__global__ void fps_kernel(const float* __restrict__ xyz, int N, int npoint,
                           float* __restrict__ dist, int* __restrict__ out)
{
    int b = blockIdx.x;
    const float* p = xyz + (size_t)b * N * 2;
    float* d = dist + (size_t)b * N;
    int tid = threadIdx.x;
    __shared__ float sv[256];
    __shared__ int   si[256];
    __shared__ int   far_s;
    for (int i = tid; i < N; i += 256) d[i] = 1e10f;
    if (tid == 0) far_s = 0;
    __syncthreads();
    for (int it = 0; it < npoint; ++it) {
        int far = far_s;
        if (tid == 0) out[(size_t)b * npoint + it] = far;
        float cx = p[(size_t)far * 2], cy = p[(size_t)far * 2 + 1];
        float best = -1.0f; int besti = 0x7fffffff;
        for (int i = tid; i < N; i += 256) {
            float dx = p[(size_t)i * 2] - cx, dy = p[(size_t)i * 2 + 1] - cy;
            float dd = dx * dx + dy * dy;
            float nd = fminf(d[i], dd);
            d[i] = nd;
            if (nd > best) { best = nd; besti = i; }
        }
        sv[tid] = best; si[tid] = besti;
        __syncthreads();
        for (int s = 128; s > 0; s >>= 1) {
            if (tid < s) {
                if (sv[tid + s] > sv[tid] ||
                    (sv[tid + s] == sv[tid] && si[tid + s] < si[tid])) {
                    sv[tid] = sv[tid + s]; si[tid] = si[tid + s];
                }
            }
            __syncthreads();
        }
        if (tid == 0) far_s = si[0];
        __syncthreads();
    }
}

__global__ void gather_xy_kernel(const float* __restrict__ xyz, const int* __restrict__ idx,
                                 int N, int S, float* __restrict__ out, int total)
{
    int t = blockIdx.x * blockDim.x + threadIdx.x;
    if (t >= total) return;
    int b = t / S;
    int n = idx[t];
    out[(size_t)t * 2]     = xyz[((size_t)b * N + n) * 2];
    out[(size_t)t * 2 + 1] = xyz[((size_t)b * N + n) * 2 + 1];
}

// ------------------------------------------------------------------
// Ball query: first K indices (ascending) with sqdist <= r^2,
// pad with first hit (or N-1 if none), matching the reference.
// ------------------------------------------------------------------
__global__ void ball_query_kernel(const float* __restrict__ xyz, const float* __restrict__ nxyz,
                                  int N, int S, int K, float r2, int* __restrict__ idx, int total)
{
    int t = blockIdx.x * blockDim.x + threadIdx.x;
    if (t >= total) return;
    int b = t / S;
    const float* p = xyz + (size_t)b * N * 2;
    float cx = nxyz[(size_t)t * 2], cy = nxyz[(size_t)t * 2 + 1];
    int* o = idx + (size_t)t * K;
    int cnt = 0, first = N - 1;
    for (int n = 0; n < N && cnt < K; ++n) {
        float dx = p[(size_t)n * 2] - cx, dy = p[(size_t)n * 2 + 1] - cy;
        if (dx * dx + dy * dy <= r2) {
            if (cnt == 0) first = n;
            o[cnt++] = n;
        }
    }
    for (int k = cnt; k < K; ++k) o[k] = first;
}

// ------------------------------------------------------------------
// Grouping -> f16, zero-padded to Cpad (so GEMM needs no K masking)
// g[b,s,k,:] = concat(points[b,idx], xyz[b,idx]-new_xyz[b,s])
// ------------------------------------------------------------------
__global__ void group_kernel(const float* __restrict__ pts, const float* __restrict__ xyz,
                             const float* __restrict__ nxyz, const int* __restrict__ idx,
                             int N, int S, int K, int Cp, int Cpad,
                             _Float16* __restrict__ g, int total)
{
    int t = blockIdx.x * blockDim.x + threadIdx.x;
    if (t >= total) return;
    int bs = t / K;
    int b = bs / S;
    int n = idx[t];
    const float* src = pts + ((size_t)b * N + n) * Cp;
    _Float16* dst = g + (size_t)t * Cpad;
    for (int c = 0; c < Cp; ++c) dst[c] = (_Float16)src[c];
    dst[Cp]     = (_Float16)(xyz[((size_t)b * N + n) * 2]     - nxyz[(size_t)bs * 2]);
    dst[Cp + 1] = (_Float16)(xyz[((size_t)b * N + n) * 2 + 1] - nxyz[(size_t)bs * 2 + 1]);
    for (int c = Cp + 2; c < Cpad; ++c) dst[c] = (_Float16)0.f;
}

// ------------------------------------------------------------------
// Fold BN into weights/bias: Wf[o, 0..lda) = W*s (f16, zero-padded),
// bf = (b-mean)*s + beta
// ------------------------------------------------------------------
__global__ void fold_kernel(const float* __restrict__ W, const float* __restrict__ b,
                            const float* __restrict__ gamma, const float* __restrict__ beta,
                            const float* __restrict__ mean, const float* __restrict__ var,
                            int O, int C, int lda, _Float16* __restrict__ Wf,
                            float* __restrict__ bf)
{
    int t = blockIdx.x * blockDim.x + threadIdx.x;
    if (t < O) {
        float s = gamma[t] * rsqrtf(var[t] + EPS_BN);
        bf[t] = (b[t] - mean[t]) * s + beta[t];
    }
    if (t < O * lda) {
        int o = t / lda, c = t % lda;
        float s = gamma[o] * rsqrtf(var[o] + EPS_BN);
        Wf[t] = (c < C) ? (_Float16)(W[(size_t)o * C + c] * s) : (_Float16)0.f;
    }
}

// ------------------------------------------------------------------
// WMMA GEMM + bias + ReLU:  Y[M,O] = relu(A[M,lda(f16)] * Wf^T + bf)
// One wave = one 16x16 tile via v_wmma_f32_16x16x32_f16.
// lda is a multiple of 32 and rows are zero-padded, so all fragment
// loads are unconditional vector loads:
//   A lane slots = one contiguous 16-half run  (k0+16*half)
//   B lane slots = one contiguous 16-half run  (k0+16*half)
// this code's own k order, shared by A and B (not the ISA table order). Tile guard is wave-uniform -> EXEC all-ones.
// ------------------------------------------------------------------
__global__ void gemm_bn_relu_wmma(const _Float16* __restrict__ A, const _Float16* __restrict__ Wf,
                                  const float* __restrict__ bf, void* __restrict__ Yv,
                                  int M, int lda, int O, int out_f32)
{
    int gwave = (blockIdx.x * blockDim.x + threadIdx.x) >> 5;
    int lane  = threadIdx.x & 31;
    int tilesN = O >> 4;
    int tiles  = (M >> 4) * tilesN;
    if (gwave >= tiles) return;          // uniform per wave
    int tm   = (gwave / tilesN) << 4;
    int tn   = (gwave % tilesN) << 4;
    int half = lane >> 4;
    int l16  = lane & 15;

    const _Float16* arow = A  + (size_t)(tm + l16) * lda + 16 * half;
    const _Float16* wrow = Wf + (size_t)(tn + l16) * lda + 16 * half;

    v8f acc = {};
    for (int k0 = 0; k0 < lda; k0 += 32) {
        union { v16h v; v8h h[2]; } au;
        au.h[0] = *(const v8h*)(arow + k0);
        au.h[1] = *(const v8h*)(arow + k0 + 8);
        v16h bb = *(const v16h*)(wrow + k0);
        acc = __builtin_amdgcn_wmma_f32_16x16x32_f16(false, au.v, false, bb,
                                                     (short)0, acc, false, false);
    }
    float bias = bf[tn + l16];
    if (out_f32) {
        float* Y = (float*)Yv;
#pragma unroll
        for (int r = 0; r < 8; ++r) {
            int m = tm + r + 8 * half;   // C/D layout: VGPR r -> M=r (+8 hi half)
            Y[(size_t)m * O + (tn + l16)] = fmaxf(acc[r] + bias, 0.f);
        }
    } else {
        _Float16* Y = (_Float16*)Yv;
#pragma unroll
        for (int r = 0; r < 8; ++r) {
            int m = tm + r + 8 * half;
            Y[(size_t)m * O + (tn + l16)] = (_Float16)fmaxf(acc[r] + bias, 0.f);
        }
    }
}

// ------------------------------------------------------------------
// Max over K samples (f16 in, fp32 out into channel slice of concat)
// ------------------------------------------------------------------
__global__ void maxpool_kernel(const _Float16* __restrict__ g, int K, int O,
                               float* __restrict__ out, int Ctot, int coff, int total)
{
    int t = blockIdx.x * blockDim.x + threadIdx.x;
    if (t >= total) return;
    int o = t % O, bs = t / O;
    const _Float16* src = g + (size_t)bs * K * O + o;
    float m = (float)src[0];
    for (int k = 1; k < K; ++k) m = fmaxf(m, (float)src[(size_t)k * O]);
    out[(size_t)bs * Ctot + coff + o] = m;
}

// ------------------------------------------------------------------
// FP interpolation: 3-NN inverse-distance, concat [points1, interp]
// -> f16 zero-padded to Cpad
// ------------------------------------------------------------------
__global__ void fp_interp_kernel(const float* __restrict__ xyz1, const float* __restrict__ xyz2,
                                 const float* __restrict__ pts2, const float* __restrict__ pts1,
                                 int N1, int N2, int C1, int C2, int Cpad,
                                 _Float16* __restrict__ out, int total)
{
    int t = blockIdx.x * blockDim.x + threadIdx.x;
    if (t >= total) return;
    int b = t / N1;
    float cx = xyz1[(size_t)t * 2], cy = xyz1[(size_t)t * 2 + 1];
    const float* p2 = xyz2 + (size_t)b * N2 * 2;
    float d0 = 1e30f, d1 = 1e30f, d2 = 1e30f;
    int   i0 = 0, i1 = 0, i2 = 0;
    for (int n = 0; n < N2; ++n) {
        float dx = p2[(size_t)n * 2] - cx, dy = p2[(size_t)n * 2 + 1] - cy;
        float d = dx * dx + dy * dy;
        if (d < d0)      { d2 = d1; i2 = i1; d1 = d0; i1 = i0; d0 = d; i0 = n; }
        else if (d < d1) { d2 = d1; i2 = i1; d1 = d; i1 = n; }
        else if (d < d2) { d2 = d; i2 = n; }
    }
    float w0 = 1.f / (fmaxf(d0, 0.f) + 1e-8f);
    float w1 = 1.f / (fmaxf(d1, 0.f) + 1e-8f);
    float w2 = 1.f / (fmaxf(d2, 0.f) + 1e-8f);
    float ws = w0 + w1 + w2;
    w0 /= ws; w1 /= ws; w2 /= ws;
    _Float16* dst = out + (size_t)t * Cpad;
    if (C1 > 0) {
        const float* s1 = pts1 + (size_t)t * C1;
        for (int c = 0; c < C1; ++c) dst[c] = (_Float16)s1[c];
    }
    const float* q = pts2 + (size_t)b * N2 * C2;
    for (int c = 0; c < C2; ++c)
        dst[C1 + c] = (_Float16)(w0 * q[(size_t)i0 * C2 + c] + w1 * q[(size_t)i1 * C2 + c]
                               + w2 * q[(size_t)i2 * C2 + c]);
    for (int c = C1 + C2; c < Cpad; ++c) dst[c] = (_Float16)0.f;
}

// ------------------------------------------------------------------
// Head prep: l0f (B*N, Cc) fp32 -> xh (B, Cc, N+1), leading zero pad
// ------------------------------------------------------------------
__global__ void headprep_kernel(const float* __restrict__ l0f, float* __restrict__ xh,
                                int B, int Cc, int N)
{
    int t = blockIdx.x * blockDim.x + threadIdx.x;
    int T = N + 1;
    int total = B * Cc * T;
    if (t >= total) return;
    int rem = t % (Cc * T);
    int b = t / (Cc * T);
    int c = rem / T;
    int tt = rem % T;
    float v = 0.f;
    if (tt > 0) v = l0f[((size_t)b * N + (tt - 1)) * Cc + c];
    xh[t] = v;
}

// ------------------------------------------------------------------
// 1D conv (cross-correlation) + optional BN+ReLU epilogue.
// x: (B,Cin,Tin), W: (O,Cin,KW), y: (B,O,Tout)
// ------------------------------------------------------------------
__global__ void conv1d_kernel(const float* __restrict__ x, const float* __restrict__ W,
                              const float* __restrict__ bias, const float* __restrict__ gamma,
                              const float* __restrict__ beta, const float* __restrict__ mean,
                              const float* __restrict__ var, int Cin, int Tin, int KW, int P,
                              int Tout, int O, float* __restrict__ y, int useBN)
{
    int b = blockIdx.z;
    int o = blockIdx.y;
    int t = blockIdx.x * blockDim.x + threadIdx.x;
    if (t >= Tout) return;
    const float* xb = x + (size_t)b * Cin * Tin;
    const float* wo = W + (size_t)o * Cin * KW;
    int i0 = P - t;        if (i0 < 0) i0 = 0;
    int i1 = Tin + P - t;  if (i1 > KW) i1 = KW;
    float acc = 0.f;
    for (int c = 0; c < Cin; ++c) {
        const float* xc = xb + (size_t)c * Tin + (t - P);
        const float* wc = wo + (size_t)c * KW;
        for (int i = i0; i < i1; ++i) acc += wc[i] * xc[i];
    }
    acc += bias[o];
    if (useBN) {
        float s = gamma[o] * rsqrtf(var[o] + EPS_BN);
        acc = fmaxf((acc - mean[o]) * s + beta[o], 0.f);
    }
    y[((size_t)b * O + o) * Tout + t] = acc;
}

// ------------------------------------------------------------------
// log_softmax over channels, write transposed (B,T,O)
// ------------------------------------------------------------------
__global__ void logsoftmax_kernel(const float* __restrict__ y, float* __restrict__ out,
                                  int T, int O, int total)
{
    int t = blockIdx.x * blockDim.x + threadIdx.x;
    if (t >= total) return;
    int b = t / T, tt = t % T;
    const float* src = y + (size_t)b * O * T + tt;
    float mx = -1e30f;
    for (int o = 0; o < O; ++o) mx = fmaxf(mx, src[(size_t)o * T]);
    float sum = 0.f;
    for (int o = 0; o < O; ++o) sum += expf(src[(size_t)o * T] - mx);
    float lse = mx + logf(sum);
    float* dst = out + (size_t)t * O;
    for (int o = 0; o < O; ++o) dst[o] = src[(size_t)o * T] - lse;
}

// l3 (B,S,C) -> out (B,C,S)
__global__ void transpose_l3_kernel(const float* __restrict__ l3, float* __restrict__ out,
                                    int S, int C, int total)
{
    int t = blockIdx.x * blockDim.x + threadIdx.x;
    if (t >= total) return;
    int b = t / (S * C);
    int rem = t % (S * C);
    int c = rem / S;
    int s = rem % S;
    out[t] = l3[((size_t)b * S + s) * C + c];
}

// ==================================================================
// Host orchestration
// ==================================================================
struct CB { const float *W, *b, *gamma, *beta, *mean, *var; };

static inline int align32(int x) { return (x + 31) & ~31; }

extern "C" void kernel_launch(void* const* d_in, const int* in_sizes, int n_in,
                              void* d_out, int out_size, void* d_ws, size_t ws_size,
                              hipStream_t stream)
{
    (void)in_sizes; (void)n_in; (void)out_size; (void)ws_size;
    const int B = 8, N0 = 2048;

    int p = 0;
    auto nextf = [&]() { return (const float*)d_in[p++]; };
    auto readCB = [&]() { CB c; c.W = nextf(); c.b = nextf(); c.gamma = nextf();
                          c.beta = nextf(); c.mean = nextf(); c.var = nextf(); return c; };

    const float* xyv = nextf();
    CB sa[3][2][3];
    for (int s = 0; s < 3; ++s)
        for (int sc = 0; sc < 2; ++sc)
            for (int l = 0; l < 3; ++l) sa[s][sc][l] = readCB();
    CB fp1[2]; for (int l = 0; l < 2; ++l) fp1[l] = readCB();
    CB fp2[2]; for (int l = 0; l < 2; ++l) fp2[l] = readCB();
    CB fp3[3]; for (int l = 0; l < 3; ++l) fp3[l] = readCB();
    const float* hW1 = nextf(); const float* hb1 = nextf();
    const float* hg1 = nextf(); const float* hbe1 = nextf();
    const float* hm1 = nextf(); const float* hv1 = nextf();
    const float* hW2 = nextf(); const float* hb2 = nextf();
    const float* hg2 = nextf(); const float* hbe2 = nextf();
    const float* hm2 = nextf(); const float* hv2 = nextf();
    const float* hW3 = nextf(); const float* hb3 = nextf();

    // ---------------- workspace carve ----------------
    char* base = (char*)d_ws;
    size_t off = 0;
    auto alloc = [&](size_t bytes) -> void* {
        void* r = base + off;
        off += (bytes + 255) & ~(size_t)255;
        return r;
    };
    float* l0_points = (float*)alloc((size_t)B * N0 * 4 * 4);
    float* l0_xy     = (float*)alloc((size_t)B * N0 * 2 * 4);
    float* l1_xy     = (float*)alloc((size_t)B * 1024 * 2 * 4);
    float* l2_xy     = (float*)alloc((size_t)B * 512 * 2 * 4);
    float* l3_xy     = (float*)alloc((size_t)B * 256 * 2 * 4);
    float* l1_pts    = (float*)alloc((size_t)B * 1024 * 192 * 4);
    float* l2_pts    = (float*)alloc((size_t)B * 512 * 192 * 4);
    float* l3_pts    = (float*)alloc((size_t)B * 256 * 256 * 4);
    float* l2_fp     = (float*)alloc((size_t)B * 512 * 256 * 4);
    float* l1_fp     = (float*)alloc((size_t)B * 1024 * 128 * 4);
    float* l0_fp     = (float*)alloc((size_t)B * N0 * 128 * 4);
    float* distbuf   = (float*)alloc((size_t)B * N0 * 4);
    int*   fpsidx    = (int*)alloc((size_t)B * 1024 * 4);
    int*   ballidx   = (int*)alloc((size_t)B * 1024 * 32 * 4);
    _Float16* Wfbuf  = (_Float16*)alloc((size_t)131072 * 2);
    float* bfbuf     = (float*)alloc((size_t)256 * 4);
    // two big generic scratch buffers (f16 for GEMM chains / fp32 for head)
    void* sb0        = alloc((size_t)33554432 * 4);   // 128 MB
    void* sb1        = alloc((size_t)33554432 * 4);   // 128 MB
    _Float16* hbufs[2] = { (_Float16*)sb0, (_Float16*)sb1 };

    const int THR = 256;
    auto blk = [&](long long n) { return dim3((unsigned)((n + THR - 1) / THR)); };

    // ---------------- layer runner (fold + WMMA GEMM) ----------------
    // A: f16 (M, lda) zero-padded; output stride O (always multiple of 32)
    auto run_layer = [&](const _Float16* A, int M, int C, int lda, const CB& cb, int O,
                         void* Y, int out_f32) {
        long long tot = (long long)O * lda;
        fold_kernel<<<blk(tot), THR, 0, stream>>>(cb.W, cb.b, cb.gamma, cb.beta, cb.mean,
                                                  cb.var, O, C, lda, Wfbuf, bfbuf);
        long long tiles = (long long)(M >> 4) * (O >> 4);
        long long blocks = (tiles + 7) / 8;   // 8 waves per 256-thread block
        gemm_bn_relu_wmma<<<dim3((unsigned)blocks), THR, 0, stream>>>(A, Wfbuf, bfbuf, Y,
                                                                      M, lda, O, out_f32);
    };

    // ---------------- SA stage runner ----------------
    auto run_sa = [&](const float* xyz, const float* pts, int N, int Cp, int S,
                      const float* radii, const int* Ks, const CB scales[2][3],
                      const int widths[2][3], float* new_xyz, float* new_pts, int Ctot) {
        fps_kernel<<<dim3(B), THR, 0, stream>>>(xyz, N, S, distbuf, fpsidx);
        gather_xy_kernel<<<blk((long long)B * S), THR, 0, stream>>>(xyz, fpsidx, N, S,
                                                                    new_xyz, B * S);
        int coff = 0;
        for (int sc = 0; sc < 2; ++sc) {
            int K = Ks[sc];
            float r2 = radii[sc] * radii[sc];
            ball_query_kernel<<<blk((long long)B * S), THR, 0, stream>>>(xyz, new_xyz, N, S, K,
                                                                         r2, ballidx, B * S);
            int M = B * S * K;
            int Cpad = align32(Cp + 2);
            group_kernel<<<blk((long long)M), THR, 0, stream>>>(pts, xyz, new_xyz, ballidx,
                                                                N, S, K, Cp, Cpad, hbufs[0], M);
            int cur = 0, C = Cp + 2, lda = Cpad;
            for (int l = 0; l < 3; ++l) {
                int O = widths[sc][l];
                run_layer(hbufs[cur], M, C, lda, scales[sc][l], O, hbufs[cur ^ 1], 0);
                cur ^= 1;
                C = O; lda = O;
            }
            int O = widths[sc][2];
            maxpool_kernel<<<blk((long long)B * S * O), THR, 0, stream>>>(hbufs[cur], K, O,
                                                                          new_pts, Ctot, coff,
                                                                          B * S * O);
            coff += O;
        }
    };

    // ---------------- FP stage runner ----------------
    auto run_fp = [&](const float* xyz1, const float* xyz2, const float* pts1,
                      const float* pts2, int N1, int N2, int C1, int C2,
                      const CB* layers, const int* widths, int nl, float* outp) {
        int Cin = C1 + C2;
        int Cpad = align32(Cin);
        fp_interp_kernel<<<blk((long long)B * N1), THR, 0, stream>>>(xyz1, xyz2, pts2, pts1,
                                                                     N1, N2, C1, C2, Cpad,
                                                                     hbufs[0], B * N1);
        const _Float16* curp = hbufs[0];
        int ci = 0, C = Cin, lda = Cpad, M = B * N1;
        for (int l = 0; l < nl; ++l) {
            int last = (l == nl - 1);
            void* Y = last ? (void*)outp : (void*)hbufs[ci ^ 1];
            run_layer(curp, M, C, lda, layers[l], widths[l], Y, last);
            curp = (const _Float16*)hbufs[ci ^ 1]; ci ^= 1;
            C = widths[l]; lda = C;
        }
    };

    // ---------------- pipeline ----------------
    prep_kernel<<<blk((long long)B * N0), THR, 0, stream>>>(xyv, l0_points, l0_xy, B, N0);

    const float r1[2] = {1.f, 3.f}; const int k1[2] = {8, 32};
    const float r2_[2] = {2.f, 4.f}; const int k2[2] = {8, 32};
    const float r3[2] = {3.f, 6.f}; const int k3[2] = {16, 32};
    const int w12[2][3] = {{32, 32, 64}, {64, 64, 128}};
    const int w3[2][3]  = {{64, 64, 128}, {64, 64, 128}};

    run_sa(l0_xy, l0_points, N0,  4,  1024, r1,  k1, sa[0], w12, l1_xy, l1_pts, 192);
    run_sa(l1_xy, l1_pts,    1024, 192, 512, r2_, k2, sa[1], w12, l2_xy, l2_pts, 192);
    run_sa(l2_xy, l2_pts,    512, 192, 256, r3,  k3, sa[2], w3,  l3_xy, l3_pts, 256);

    const int wfp1[2] = {256, 256};
    const int wfp2[2] = {128, 128};
    const int wfp3[3] = {128, 128, 128};
    run_fp(l2_xy, l3_xy, nullptr, l3_pts, 512, 256, 0,   256, fp1, wfp1, 2, l2_fp);
    run_fp(l1_xy, l2_xy, l1_pts,  l2_fp, 1024, 512, 192, 256, fp2, wfp2, 2, l1_fp);
    run_fp(l0_xy, l1_xy, l0_points, l1_fp, N0, 1024, 4,  128, fp3, wfp3, 3, l0_fp);

    // ---------------- head (reuse big scratch as fp32) ----------------
    float* fb0 = (float*)sb0;
    float* fb1 = (float*)sb1;
    float* xhead = fb0;                          // (B,128,2049)
    float* y1    = fb0 + (size_t)B * 128 * 2049; // (B,128,2048)
    float* y2    = fb1;                          // (B,128,2047)
    float* y3    = fb1 + (size_t)B * 128 * 2047; // (B,5,2048)

    headprep_kernel<<<blk((long long)B * 128 * 2049), THR, 0, stream>>>(l0_fp, xhead, B, 128, N0);

    conv1d_kernel<<<dim3((2048 + THR - 1) / THR, 128, B), THR, 0, stream>>>(
        xhead, hW1, hb1, hg1, hbe1, hm1, hv1, 128, 2049, 256, 127, 2048, 128, y1, 1);
    conv1d_kernel<<<dim3((2047 + THR - 1) / THR, 128, B), THR, 0, stream>>>(
        y1, hW2, hb2, hg2, hbe2, hm2, hv2, 128, 2048, 128, 63, 2047, 128, y2, 1);
    conv1d_kernel<<<dim3((2048 + THR - 1) / THR, 5, B), THR, 0, stream>>>(
        y2, hW3, hb3, nullptr, nullptr, nullptr, nullptr, 128, 2047, 6, 3, 2048, 5, y3, 0);

    float* out = (float*)d_out;
    logsoftmax_kernel<<<blk((long long)B * 2048), THR, 0, stream>>>(y3, out, 2048, 5, B * 2048);
    transpose_l3_kernel<<<blk((long long)B * 256 * 256), THR, 0, stream>>>(
        l3_pts, out + (size_t)B * 2048 * 5, 256, 256, B * 256 * 256);
}
